// VolumeRenderer_249108103672
// MI455X (gfx1250) — matrix operand order repaired and checked by reading, not compiled, not run on hardware
//
#include <hip/hip_runtime.h>
#include <hip/hip_bf16.h>

typedef __attribute__((ext_vector_type(16))) _Float16 v16h;
typedef __attribute__((ext_vector_type(8)))  float    v8f;

#define NERF_B   8192
#define NC       64
#define NF       128
#define NT       192          // NC + NF samples in fine pass
#define HID      128
#define T_NEAR   2.0f
#define T_FAR    6.0f
#define PERSIST_BLOCKS 4096   // persistent MLP blocks (tile-strided)

// ---------------------------------------------------------------------------
// 1. Stratified coarse sampling: t_c [B,NC], pts [B*NC,3]
// ---------------------------------------------------------------------------
__global__ __launch_bounds__(256)
void gen_coarse_kernel(const float* __restrict__ rays_o,
                       const float* __restrict__ rays_d,
                       const float* __restrict__ rand_c,
                       float* __restrict__ t_c,
                       float* __restrict__ pts)
{
    int idx = blockIdx.x * 256 + threadIdx.x;
    if (idx >= NERF_B * NC) return;
    int ray = idx >> 6;
    int i   = idx & 63;
    const float step = (T_FAR - T_NEAR) / (float)(NC - 1);
    float ti    = T_NEAR + step * (float)i;
    float lower = (i == 0)      ? ti : ti - 0.5f * step;
    float upper = (i == NC - 1) ? ti : ti + 0.5f * step;
    float tt = lower + (upper - lower) * rand_c[idx];
    t_c[idx] = tt;
    float ox = rays_o[ray*3+0], oy = rays_o[ray*3+1], oz = rays_o[ray*3+2];
    float dx = rays_d[ray*3+0], dy = rays_d[ray*3+1], dz = rays_d[ray*3+2];
    pts[(size_t)idx*3+0] = ox + dx*tt;
    pts[(size_t)idx*3+1] = oy + dy*tt;
    pts[(size_t)idx*3+2] = oz + dz*tt;
}

// ---------------------------------------------------------------------------
// 2. Persistent fused MLP: 64 points per tile, WMMA f16 for the 128x128 layer.
//    Weights staged to LDS once per block; B fragments resident in VGPRs.
// ---------------------------------------------------------------------------
__global__ __launch_bounds__(256)
void mlp_wmma_kernel(const float* __restrict__ pts,     // [Ntot,3]
                     const float* __restrict__ rays_d,  // [B,3]
                     const float* __restrict__ W1, const float* __restrict__ b1,
                     const float* __restrict__ W2, const float* __restrict__ b2,
                     const float* __restrict__ Wsig, const float* __restrict__ bsig,
                     const float* __restrict__ Wrgb, const float* __restrict__ brgb,
                     float* __restrict__ rgb_out,       // [Ntot,3]
                     float* __restrict__ sigma_out,     // [Ntot]
                     int n_per_ray, int n_tiles)
{
    __shared__ _Float16 sW2t[HID * HID];     // W2 transposed: [n][k], f16
    __shared__ _Float16 sH[64 * HID];        // activations (reused in/out)
    __shared__ float    sW1[3 * HID];
    __shared__ float    sB1[HID];
    __shared__ float    sB2[HID];
    __shared__ float    sWsig[HID];
    __shared__ float    sWrgb[(HID + 3) * 3];
    __shared__ float    sBias4[4];           // bsig, brgb[0..2]

    const int tid = threadIdx.x;

    // ---- stage all weights once per persistent block -----------------------
    for (int i = tid; i < HID * HID; i += 256) {
        int k = i >> 7, n = i & (HID - 1);
        sW2t[n * HID + k] = (_Float16)W2[i];
    }
    for (int i = tid; i < 3 * HID; i += 256) sW1[i] = W1[i];
    for (int i = tid; i < (HID + 3) * 3; i += 256) sWrgb[i] = Wrgb[i];
    if (tid < HID) { sB1[tid] = b1[tid]; sB2[tid] = b2[tid]; sWsig[tid] = Wsig[tid]; }
    if (tid == 0) { sBias4[0] = bsig[0]; sBias4[1] = brgb[0]; sBias4[2] = brgb[1]; sBias4[3] = brgb[2]; }
    __syncthreads();

    // ---- hoist tile-invariant B fragments of W2 into registers -------------
    const int wave = tid >> 5;
    const int lane = tid & 31;
    const int ln   = lane & 15;            // row (A) / col (B) within tile
    const int hi   = lane >> 4;            // half-select per ISA layout
    const int rt   = wave & 3;             // row tile: rows rt*16..rt*16+15
    const int nh   = wave >> 2;            // n half:   cols nh*64..nh*64+63

    v16h Bfrag[16];                        // [kk*4 + q]
    #pragma unroll
    for (int kk = 0; kk < 4; ++kk) {
        #pragma unroll
        for (int q = 0; q < 4; ++q) {
            int n = (nh * 4 + q) * 16 + ln;
            const _Float16* br = &sW2t[n * HID + kk * 32 + 16 * hi];
            #pragma unroll
            for (int j = 0; j < 16; ++j) Bfrag[kk * 4 + q][j] = br[j];
        }
    }

    // ---- persistent tile loop ----------------------------------------------
    for (int tile = blockIdx.x; tile < n_tiles; tile += gridDim.x) {
        const int block0 = tile * 64;

        // layer 1: h = relu(pts @ W1 + b1) (K=3, VALU)
        #pragma unroll
        for (int i = 0; i < 32; ++i) {
            int idx = i * 256 + tid;       // 0 .. 8191
            int p = idx >> 7, c = idx & (HID - 1);
            const float* P = &pts[(size_t)(block0 + p) * 3];
            float h = P[0]*sW1[c] + P[1]*sW1[HID+c] + P[2]*sW1[2*HID+c] + sB1[c];
            sH[p * HID + c] = (_Float16)fmaxf(h, 0.0f);
        }
        // speculative prefetch of next tile's points
        {
            int nb0 = block0 + (int)gridDim.x * 64;
            if (tid < 64) __builtin_prefetch(&pts[(size_t)(nb0 + tid) * 3], 0, 0);
        }
        __syncthreads();

        // layer 2: h = relu(h @ W2 + b2) via v_wmma_f32_16x16x32_f16
        v8f acc[4];
        #pragma unroll
        for (int q = 0; q < 4; ++q)
            #pragma unroll
            for (int e = 0; e < 8; ++e) acc[q][e] = 0.0f;

        #pragma unroll
        for (int kk = 0; kk < 4; ++kk) {
            // A fragment: row = rt*16+ln ; k = 32*kk + 16*hi + 0..15
            v16h a;
            const _Float16* ar = &sH[(rt * 16 + ln) * HID + kk * 32];
            #pragma unroll
            for (int j = 0; j < 16; ++j) {
                a[j] = ar[16 * hi + j];
            }
            #pragma unroll
            for (int q = 0; q < 4; ++q) {
                acc[q] = __builtin_amdgcn_wmma_f32_16x16x32_f16(
                    false, a, false, Bfrag[kk * 4 + q], (short)0, acc[q],
                    false, false);
            }
        }
        __syncthreads();                   // all reads of sH done

        // bias + relu, write back into sH (f16)
        #pragma unroll
        for (int q = 0; q < 4; ++q) {
            int col = (nh * 4 + q) * 16 + ln;
            #pragma unroll
            for (int i = 0; i < 8; ++i) {
                int row = rt * 16 + i + 8 * hi;
                float v = acc[q][i] + sB2[col];
                sH[row * HID + col] = (_Float16)fmaxf(v, 0.0f);
            }
        }
        __syncthreads();

        // heads: sigma (threads 0..63), rgb (threads 64..255)
        if (tid < 64) {
            int p = tid;
            const _Float16* hrow = &sH[p * HID];
            float s = sBias4[0];
            for (int j = 0; j < HID; ++j) {
                int k = (j + p) & (HID - 1);   // rotated start: spread banks
                s += (float)hrow[k] * sWsig[k];
            }
            sigma_out[block0 + p] = fmaxf(s, 0.0f);
        } else {
            int q = tid - 64;                  // 0..191
            int p = q / 3, c = q % 3;
            size_t gidx = (size_t)block0 + p;
            int ray = (int)(gidx / (size_t)n_per_ray);
            const _Float16* hrow = &sH[p * HID];
            float o = sBias4[1 + c];
            for (int j = 0; j < HID; ++j) {
                int k = (j + p) & (HID - 1);
                o += (float)hrow[k] * sWrgb[k * 3 + c];
            }
            const float* D = &rays_d[ray * 3];
            o += D[0] * sWrgb[(HID + 0) * 3 + c]
               + D[1] * sWrgb[(HID + 1) * 3 + c]
               + D[2] * sWrgb[(HID + 2) * 3 + c];
            rgb_out[gidx * 3 + c] = 1.0f / (1.0f + __expf(-o));
        }
        __syncthreads();                   // sH reused by next tile's layer 1
    }
}

// ---------------------------------------------------------------------------
// 3. Volume rendering: one thread per ray (sequential transmittance scan)
// ---------------------------------------------------------------------------
__global__ __launch_bounds__(256)
void render_kernel(const float* __restrict__ sigma,
                   const float* __restrict__ rgb,
                   const float* __restrict__ t,
                   int N,
                   float* __restrict__ color_out,
                   float* __restrict__ depth_out,
                   float* __restrict__ weights_out)
{
    int ray = blockIdx.x * blockDim.x + threadIdx.x;
    if (ray >= NERF_B) return;
    const float* ts = t     + (size_t)ray * N;
    const float* sg = sigma + (size_t)ray * N;
    const float* rg = rgb   + (size_t)ray * N * 3;
    float T = 1.0f, c0 = 0.f, c1 = 0.f, c2 = 0.f, dep = 0.f;
    for (int i = 0; i < N; ++i) {
        float ti    = ts[i];
        float delta = (i + 1 < N) ? (ts[i + 1] - ti) : 1e10f;
        float alpha = 1.0f - __expf(-sg[i] * delta);
        float w = T * alpha;
        T *= (1.0f - alpha + 1e-10f);
        c0 += w * rg[i * 3 + 0];
        c1 += w * rg[i * 3 + 1];
        c2 += w * rg[i * 3 + 2];
        dep += w * ti;
        if (weights_out) weights_out[(size_t)ray * N + i] = w;
    }
    color_out[ray * 3 + 0] = c0;
    color_out[ray * 3 + 1] = c1;
    color_out[ray * 3 + 2] = c2;
    if (depth_out) depth_out[ray] = dep;
}

// ---------------------------------------------------------------------------
// 4. Importance sampling: one block per ray
// ---------------------------------------------------------------------------
__global__ __launch_bounds__(256)
void importance_kernel(const float* __restrict__ rays_o,
                       const float* __restrict__ rays_d,
                       const float* __restrict__ t_c,
                       const float* __restrict__ weights_c,
                       const float* __restrict__ u_fine,
                       float* __restrict__ t_f,       // [B,NT]
                       float* __restrict__ pts_f)     // [B*NT,3]
{
    __shared__ float s_cdf[NC + 1];
    __shared__ float s_tc[NC];
    __shared__ float s_all[NT];
    __shared__ float s_sorted[NT];

    int ray = blockIdx.x;
    int tid = threadIdx.x;

    if (tid < NC) s_tc[tid] = t_c[(size_t)ray * NC + tid];
    __syncthreads();

    if (tid == 0) {
        float w[NC]; float sum = 0.f;
        for (int i = 0; i < NC; ++i) {
            w[i] = weights_c[(size_t)ray * NC + i] + 1e-5f;
            sum += w[i];
        }
        float inv = 1.0f / sum, c = 0.f;
        s_cdf[0] = 0.f;
        for (int i = 0; i < NC; ++i) { c += w[i] * inv; s_cdf[i + 1] = c; }
    }
    __syncthreads();

    if (tid < NF) {
        float u = u_fine[(size_t)ray * NF + tid];
        // searchsorted(cdf, u, side='right'): first idx with cdf[idx] > u
        int lo = 0, hh = NC + 1;
        while (lo < hh) {
            int mid = (lo + hh) >> 1;
            if (s_cdf[mid] <= u) lo = mid + 1; else hh = mid;
        }
        int below = min(max(lo - 1, 0), NC - 1);
        int above = min(max(lo,     0), NC - 1);
        s_all[NC + tid] = 0.5f * (s_tc[below] + s_tc[above]);
    }
    if (tid < NC) s_all[tid] = s_tc[tid];
    __syncthreads();

    // rank-based sort of the 192 merged t values
    if (tid < NT) {
        float ti = s_all[tid];
        int r = 0;
        for (int j = 0; j < NT; ++j) {
            float tj = s_all[j];
            r += (tj < ti) || (tj == ti && j < tid);
        }
        s_sorted[r] = ti;
    }
    __syncthreads();

    if (tid < NT) {
        float tt = s_sorted[tid];
        size_t gi = (size_t)ray * NT + tid;
        t_f[gi] = tt;
        float ox = rays_o[ray*3+0], oy = rays_o[ray*3+1], oz = rays_o[ray*3+2];
        float dx = rays_d[ray*3+0], dy = rays_d[ray*3+1], dz = rays_d[ray*3+2];
        pts_f[gi*3+0] = ox + dx*tt;
        pts_f[gi*3+1] = oy + dy*tt;
        pts_f[gi*3+2] = oz + dz*tt;
    }
}

// ---------------------------------------------------------------------------
// Launcher
// ---------------------------------------------------------------------------
extern "C" void kernel_launch(void* const* d_in, const int* in_sizes, int n_in,
                              void* d_out, int out_size, void* d_ws, size_t ws_size,
                              hipStream_t stream)
{
    (void)in_sizes; (void)n_in; (void)out_size; (void)ws_size;
    const float* rays_o = (const float*)d_in[0];
    const float* rays_d = (const float*)d_in[1];
    const float* rand_c = (const float*)d_in[2];
    const float* u_fine = (const float*)d_in[3];
    const float* W1   = (const float*)d_in[4];
    const float* b1   = (const float*)d_in[5];
    const float* W2   = (const float*)d_in[6];
    const float* b2   = (const float*)d_in[7];
    const float* Wsig = (const float*)d_in[8];
    const float* bsig = (const float*)d_in[9];
    const float* Wrgb = (const float*)d_in[10];
    const float* brgb = (const float*)d_in[11];
    float* out = (float*)d_out;

    float* ws  = (float*)d_ws;
    float* t_c = ws;                                  // B*NC
    float* t_f = t_c + (size_t)NERF_B * NC;           // B*NT
    float* pts = t_f + (size_t)NERF_B * NT;           // B*NT*3 (coarse uses prefix)
    float* rgb = pts + (size_t)NERF_B * NT * 3;       // B*NT*3
    float* sig = rgb + (size_t)NERF_B * NT * 3;       // B*NT
    float* wts = sig + (size_t)NERF_B * NT;           // B*NC

    float* color_c = out;                             // [B,3]
    float* color_f = out + (size_t)NERF_B * 3;        // [B,3]
    float* depth_f = out + (size_t)NERF_B * 6;        // [B]

    const int coarse_tiles = NERF_B * NC / 64;        // 8192
    const int fine_tiles   = NERF_B * NT / 64;        // 24576
    const int gridc = (coarse_tiles < PERSIST_BLOCKS) ? coarse_tiles : PERSIST_BLOCKS;
    const int gridf = (fine_tiles   < PERSIST_BLOCKS) ? fine_tiles   : PERSIST_BLOCKS;

    // coarse pass
    gen_coarse_kernel<<<NERF_B * NC / 256, 256, 0, stream>>>(
        rays_o, rays_d, rand_c, t_c, pts);
    mlp_wmma_kernel<<<gridc, 256, 0, stream>>>(
        pts, rays_d, W1, b1, W2, b2, Wsig, bsig, Wrgb, brgb, rgb, sig,
        NC, coarse_tiles);
    render_kernel<<<(NERF_B + 255) / 256, 256, 0, stream>>>(
        sig, rgb, t_c, NC, color_c, nullptr, wts);

    // importance resampling
    importance_kernel<<<NERF_B, 256, 0, stream>>>(
        rays_o, rays_d, t_c, wts, u_fine, t_f, pts);

    // fine pass
    mlp_wmma_kernel<<<gridf, 256, 0, stream>>>(
        pts, rays_d, W1, b1, W2, b2, Wsig, bsig, Wrgb, brgb, rgb, sig,
        NT, fine_tiles);
    render_kernel<<<(NERF_B + 255) / 256, 256, 0, stream>>>(
        sig, rgb, t_f, NT, color_f, depth_f, nullptr);
}
